// SparseDSAAttention_76768245449376
// MI455X (gfx1250) — matrix operand order repaired and checked by reading, not compiled, not run on hardware
//
#include <hip/hip_runtime.h>

// ---------------------------------------------------------------------------
// Problem constants (match reference)
// ---------------------------------------------------------------------------
constexpr int kT    = 2048;
constexpr int kHID  = 4096;
constexpr int kH    = 32;
constexpr int kKVH  = 8;
constexpr int kD    = 128;
constexpr int kTOPK = 256;
constexpr int kSINK = 16;
constexpr int kWIN  = 256;
constexpr float kSCALE = 0.088388347648318447f;  // 1/sqrt(128)
constexpr float kEPS   = 1e-6f;

// ---------------------------------------------------------------------------
// WMMA helper types (wave32, gfx1250)
// ---------------------------------------------------------------------------
typedef _Float16 h8   __attribute__((ext_vector_type(8)));
typedef _Float16 v16h __attribute__((ext_vector_type(16)));
typedef float    v8f  __attribute__((ext_vector_type(8)));
typedef unsigned int u32x4 __attribute__((ext_vector_type(4)));
typedef int          i32x8 __attribute__((ext_vector_type(8)));
typedef int          i32x4 __attribute__((ext_vector_type(4)));

#if __has_builtin(__builtin_amdgcn_tensor_load_to_lds)
#define HAS_TDM 1
#else
#define HAS_TDM 0
#endif

__device__ __forceinline__ v16h shuf16(h8 lo, h8 hi) {
  return __builtin_shufflevector(lo, hi, 0,1,2,3,4,5,6,7,8,9,10,11,12,13,14,15);
}
__device__ __forceinline__ v8f vzero8() {
  v8f z;
  #pragma unroll
  for (int i = 0; i < 8; ++i) z[i] = 0.f;
  return z;
}
// D = A(16x32 f16) * B(32x16 f16) + C(16x16 f32)
__device__ __forceinline__ v8f wmma_f16(v16h a, v16h b, v8f c) {
  return __builtin_amdgcn_wmma_f32_16x16x32_f16(false, a, false, b,
                                                (short)0, c, false, false);
}

// ---------------------------------------------------------------------------
// Kernel 1/5: fp32 (or f16) A  x  fp32 B  ->  fp32 C  via f16 WMMA, f32 accum
//   Block tile 128x128, K-step 32. 256 threads = 8 waves, each wave owns a
//   32x64 sub-tile (2x4 WMMA accumulators).
//   A_HALF path: A tile (128x32 f16, row-major, no conversion needed) is
//   staged by the Tensor Data Mover (tensor_load_to_lds, TENSORcnt), freeing
//   VALU/VMEM slots for the fp32->f16 B staging + WMMA stream.
// ---------------------------------------------------------------------------
template <bool A_HALF>
__global__ __launch_bounds__(256)
void gemm_wmma(const void* __restrict__ Ap, const float* __restrict__ B,
               float* __restrict__ C, int M, int N, int K)
{
  __shared__ _Float16 As[128][32];   // 8 KB
  __shared__ _Float16 Bs[128][32];   // 8 KB, [n][k]

  const int tid  = threadIdx.x;
  const int wave = tid >> 5;
  const int lane = tid & 31;
  const int ml   = lane & 15;
  const int m0   = blockIdx.y * 128;
  const int n0   = blockIdx.x * 128;
  const int wm   = (wave >> 1) * 32;   // 4 wave-rows
  const int wn   = (wave & 1) * 64;    // 2 wave-cols
  const int kb_a = (lane < 16) ? 0 : 8;
  const int kb_b = (lane < 16) ? 0 : 8;

  v8f acc[2][4];
  #pragma unroll
  for (int mi = 0; mi < 2; ++mi)
    #pragma unroll
    for (int ni = 0; ni < 4; ++ni) acc[mi][ni] = vzero8();

  for (int k0 = 0; k0 < K; k0 += 32) {
    __syncthreads();
    // ---- stage A tile (128x32) -> f16 LDS --------------------------------
    if constexpr (A_HALF) {
#if HAS_TDM
      // TDM: DMA the row-major f16 tile straight into LDS (one issue, wave 0)
      if (wave == 0) {
        const unsigned lds_off = (unsigned)(size_t)(void*)&As[0][0];
        const unsigned long long ga =
            (unsigned long long)(const _Float16*)Ap +
            ((unsigned long long)m0 * (unsigned)K + (unsigned)k0) * 2ull;
        u32x4 g0;
        g0[0] = 1u;                                    // count=1, user mode
        g0[1] = lds_off;                               // lds_addr (bytes)
        g0[2] = (unsigned)(ga & 0xffffffffull);        // global_addr lo
        g0[3] = (unsigned)((ga >> 32) & 0x01ffffffull) // global_addr hi
                | (2u << 30);                          // type = 2 ("image")
        i32x8 g1;
        g1[0] = 0x00010000;                            // data_size = 1 (2B)
        g1[1] = (int)(((unsigned)K & 0xffffu) << 16);  // tensor_dim0 lo16
        g1[2] = (int)(((unsigned)K >> 16) |
                      (((unsigned)M & 0xffffu) << 16));// dim0 hi | dim1 lo
        g1[3] = (int)(((unsigned)M >> 16) |
                      (32u << 16));                    // dim1 hi | tile_dim0
        g1[4] = 128;                                   // tile_dim1 = 128 rows
        g1[5] = K;                                     // tensor_dim0_stride
        g1[6] = 0;
        g1[7] = 0;
        i32x4 gz;
        gz[0] = 0; gz[1] = 0; gz[2] = 0; gz[3] = 0;
        i32x8 gz8;
        #pragma unroll
        for (int i = 0; i < 8; ++i) gz8[i] = 0;
        __builtin_amdgcn_tensor_load_to_lds(g0, g1, gz, gz, gz8, 0);
        __builtin_amdgcn_s_wait_tensorcnt(0);
      }
#else
      {
        const int r  = tid >> 1;
        const int kk = (tid & 1) * 16;
        const _Float16* src = (const _Float16*)Ap + (size_t)(m0 + r) * K + k0 + kk;
        *(h8*)&As[r][kk]     = *(const h8*)(src);
        *(h8*)&As[r][kk + 8] = *(const h8*)(src + 8);
        if (k0 + 32 < K) __builtin_prefetch(src + 32, 0, 0);
      }
#endif
    } else {
      const int r  = tid >> 1;
      const int kk = (tid & 1) * 16;
      const float* src = (const float*)Ap + (size_t)(m0 + r) * K + k0 + kk;
      #pragma unroll
      for (int j = 0; j < 16; ++j) As[r][kk + j] = (_Float16)src[j];
      if (k0 + 32 < K) __builtin_prefetch(src + 32, 0, 0);
    }
    // ---- stage B tile transposed: Bs[n][k] = B[k0+k][n0+n] ---------------
    {
      const int kk = tid >> 3;             // 0..31
      const int nn = (tid & 7) * 16;       // 0..112
      const float* src = B + (size_t)(k0 + kk) * N + n0 + nn;
      #pragma unroll
      for (int j = 0; j < 16; ++j) Bs[nn + j][kk] = (_Float16)src[j];
      if (k0 + 32 < K) __builtin_prefetch(src + (size_t)32 * N, 0, 0);
    }
    __syncthreads();

    // ---- fragments + 8 WMMA ---------------------------------------------
    v16h af[2], bf[4];
    #pragma unroll
    for (int mi = 0; mi < 2; ++mi) {
      const _Float16* p = &As[wm + mi * 16 + ml][0];
      af[mi] = shuf16(*(const h8*)(p + kb_a), *(const h8*)(p + 16 + kb_a));
    }
    #pragma unroll
    for (int ni = 0; ni < 4; ++ni) {
      const _Float16* p = &Bs[wn + ni * 16 + ml][0];
      bf[ni] = shuf16(*(const h8*)(p + kb_b), *(const h8*)(p + 16 + kb_b));
    }
    #pragma unroll
    for (int mi = 0; mi < 2; ++mi)
      #pragma unroll
      for (int ni = 0; ni < 4; ++ni)
        acc[mi][ni] = wmma_f16(af[mi], bf[ni], acc[mi][ni]);
  }

  // ---- store C (C/D layout: M = r + 8*(lane>=16), N = lane&15) -----------
  const int rhi = (lane < 16) ? 0 : 8;
  #pragma unroll
  for (int mi = 0; mi < 2; ++mi)
    #pragma unroll
    for (int ni = 0; ni < 4; ++ni)
      #pragma unroll
      for (int r = 0; r < 8; ++r) {
        const int row = m0 + wm + mi * 16 + r + rhi;
        const int col = n0 + wn + ni * 16 + ml;
        C[(size_t)row * N + col] = acc[mi][ni][r];
      }
}

// ---------------------------------------------------------------------------
// Kernel 2/5: per-head RMSNorm + RoPE + f16 cast + head-major relayout.
// ---------------------------------------------------------------------------
__global__ __launch_bounds__(256)
void rmsnorm_rope(const float* __restrict__ qf, const float* __restrict__ kvf,
                  const float* __restrict__ qw, const float* __restrict__ kw,
                  const float* __restrict__ cosb, const float* __restrict__ sinb,
                  _Float16* __restrict__ qh, _Float16* __restrict__ kh,
                  _Float16* __restrict__ vh)
{
  const int t    = blockIdx.x;
  const int wave = threadIdx.x >> 5;
  const int lane = threadIdx.x & 31;

  const float c0 = cosb[t * kD + lane],      c1 = cosb[t * kD + lane + 32];
  const float c2 = cosb[t * kD + lane + 64], c3 = cosb[t * kD + lane + 96];
  const float s0 = sinb[t * kD + lane],      s1 = sinb[t * kD + lane + 32];
  const float s2 = sinb[t * kD + lane + 64], s3 = sinb[t * kD + lane + 96];

  // ---- q heads -----------------------------------------------------------
  for (int hh = wave; hh < kH; hh += 8) {
    const float* x = qf + (size_t)t * (kH * kD) + hh * kD;
    float x0 = x[lane], x1 = x[lane + 32], x2 = x[lane + 64], x3 = x[lane + 96];
    float ss = x0 * x0 + x1 * x1 + x2 * x2 + x3 * x3;
    #pragma unroll
    for (int off = 16; off; off >>= 1) ss += __shfl_xor(ss, off, 32);
    const float r = rsqrtf(ss * (1.f / kD) + kEPS);
    const float n0 = x0 * r * qw[lane],      n1 = x1 * r * qw[lane + 32];
    const float n2 = x2 * r * qw[lane + 64], n3 = x3 * r * qw[lane + 96];
    _Float16* dst = qh + ((size_t)hh * kT + t) * kD;
    dst[lane]      = (_Float16)(n0 * c0 - n2 * s0);
    dst[lane + 32] = (_Float16)(n1 * c1 - n3 * s1);
    dst[lane + 64] = (_Float16)(n2 * c2 + n0 * s2);
    dst[lane + 96] = (_Float16)(n3 * c3 + n1 * s3);
  }
  // ---- kv head (k: norm+rope, v: raw) ------------------------------------
  {
    const int g = wave;  // 8 waves == 8 kv heads
    const float* x = kvf + (size_t)t * (kKVH * kD) + g * kD;
    float x0 = x[lane], x1 = x[lane + 32], x2 = x[lane + 64], x3 = x[lane + 96];
    _Float16* dv = vh + ((size_t)g * kT + t) * kD;
    dv[lane]      = (_Float16)x0; dv[lane + 32] = (_Float16)x1;
    dv[lane + 64] = (_Float16)x2; dv[lane + 96] = (_Float16)x3;
    float ss = x0 * x0 + x1 * x1 + x2 * x2 + x3 * x3;
    #pragma unroll
    for (int off = 16; off; off >>= 1) ss += __shfl_xor(ss, off, 32);
    const float r = rsqrtf(ss * (1.f / kD) + kEPS);
    const float n0 = x0 * r * kw[lane],      n1 = x1 * r * kw[lane + 32];
    const float n2 = x2 * r * kw[lane + 64], n3 = x3 * r * kw[lane + 96];
    _Float16* dk = kh + ((size_t)g * kT + t) * kD;
    dk[lane]      = (_Float16)(n0 * c0 - n2 * s0);
    dk[lane + 32] = (_Float16)(n1 * c1 - n3 * s1);
    dk[lane + 64] = (_Float16)(n2 * c2 + n0 * s2);
    dk[lane + 96] = (_Float16)(n3 * c3 + n1 * s3);
  }
}

// ---------------------------------------------------------------------------
// Kernel 3/5: sparse attention. One block per (16-query tile, head).
//   Phase 1: S = Q Kt via WMMA, full 16x2048 row in LDS (f16, 64 KB).
//   Phase 2: top-k threshold via 24-step value bisection on the UNMASKED
//            scores (matches reference), then sink|window|topk & causal
//            mask, masked softmax; probs overwrite scores in place.
//   Phase 3: O = P V via WMMA, V tiles transposed through 8 KB LDS.
// ---------------------------------------------------------------------------
__global__ __launch_bounds__(256)
void attn_sparse(const _Float16* __restrict__ qh, const _Float16* __restrict__ kh,
                 const _Float16* __restrict__ vh, _Float16* __restrict__ aoh)
{
  __shared__ _Float16 sc[16][kT];     // 64 KB: scores, then probabilities
  __shared__ _Float16 vt[kD][32];     //  8 KB: V tile transposed [d][s]
  __shared__ float    rsum[16];

  const int h    = blockIdx.y;
  const int m0   = blockIdx.x * 16;
  const int g    = h >> 2;            // kv head (NREP = 4)
  const int wave = threadIdx.x >> 5;
  const int lane = threadIdx.x & 31;
  const int ml   = lane & 15;
  const int kb_a = (lane < 16) ? 0 : 8;
  const int kb_b = (lane < 16) ? 0 : 8;
  const int rhi  = (lane < 16) ? 0 : 8;

  // ---- preload Q A-fragments (16 rows x 128, 4 K-steps) ------------------
  v16h aq[4];
  {
    const _Float16* qrow = qh + ((size_t)h * kT + m0 + ml) * kD;
    #pragma unroll
    for (int ks = 0; ks < 4; ++ks)
      aq[ks] = shuf16(*(const h8*)(qrow + ks * 32 + kb_a),
                      *(const h8*)(qrow + ks * 32 + 16 + kb_a));
  }

  // ---- Phase 1: raw scores -----------------------------------------------
  const _Float16* kbase = kh + (size_t)g * kT * kD;
  for (int iter = 0; iter < 16; ++iter) {
    const int s0 = iter * 128 + wave * 16;
    v8f acc = vzero8();
    const _Float16* krow = kbase + (size_t)(s0 + ml) * kD;
    #pragma unroll
    for (int ks = 0; ks < 4; ++ks) {
      v16h bf = shuf16(*(const h8*)(krow + ks * 32 + kb_b),
                       *(const h8*)(krow + ks * 32 + 16 + kb_b));
      acc = wmma_f16(aq[ks], bf, acc);
    }
    #pragma unroll
    for (int r = 0; r < 8; ++r)
      sc[r + rhi][s0 + ml] = (_Float16)(acc[r] * kSCALE);
  }
  __syncthreads();

  // ---- Phase 2: top-k threshold + masked softmax (2 rows / wave) ---------
  for (int mr = 0; mr < 2; ++mr) {
    const int m  = wave * 2 + mr;
    const int tq = m0 + m;
    float rv[64];
    #pragma unroll
    for (int j = 0; j < 64; ++j) rv[j] = (float)sc[m][j * 32 + lane];

    float hiv = -1e30f, lov = 1e30f;
    #pragma unroll
    for (int j = 0; j < 64; ++j) { hiv = fmaxf(hiv, rv[j]); lov = fminf(lov, rv[j]); }
    #pragma unroll
    for (int off = 16; off; off >>= 1) {
      hiv = fmaxf(hiv, __shfl_xor(hiv, off, 32));
      lov = fminf(lov, __shfl_xor(lov, off, 32));
    }
    // value bisection: largest thr with count(score >= thr) >= TOPK
    for (int it = 0; it < 24; ++it) {
      const float mid = 0.5f * (hiv + lov);
      int cnt = 0;
      #pragma unroll
      for (int j = 0; j < 64; ++j) cnt += (rv[j] >= mid) ? 1 : 0;
      #pragma unroll
      for (int off = 16; off; off >>= 1) cnt += __shfl_xor(cnt, off, 32);
      if (cnt >= kTOPK) lov = mid; else hiv = mid;
    }
    const float thr = lov;

    float mmax = -1e30f;
    #pragma unroll
    for (int j = 0; j < 64; ++j) {
      const int s  = j * 32 + lane;
      const int dd = tq - s;
      const bool keep = ((s < kSINK) | ((dd <= kWIN) & (dd >= -kWIN)) |
                         (rv[j] >= thr)) & (s <= tq);
      if (keep) mmax = fmaxf(mmax, rv[j]);
    }
    #pragma unroll
    for (int off = 16; off; off >>= 1) mmax = fmaxf(mmax, __shfl_xor(mmax, off, 32));

    float sum = 0.f;
    #pragma unroll
    for (int j = 0; j < 64; ++j) {
      const int s  = j * 32 + lane;
      const int dd = tq - s;
      const bool keep = ((s < kSINK) | ((dd <= kWIN) & (dd >= -kWIN)) |
                         (rv[j] >= thr)) & (s <= tq);
      const float p = keep ? __expf(rv[j] - mmax) : 0.f;
      sum += p;
      sc[m][s] = (_Float16)p;           // overwrite score with prob
    }
    #pragma unroll
    for (int off = 16; off; off >>= 1) sum += __shfl_xor(sum, off, 32);
    if (lane == 0) rsum[m] = sum;
  }
  __syncthreads();

  // ---- Phase 3: O = P V ---------------------------------------------------
  v8f oacc = vzero8();
  const int n0 = wave * 16;
  const _Float16* vbase = vh + (size_t)g * kT * kD;
  for (int ks = 0; ks < kT / 32; ++ks) {
    __syncthreads();
    {   // stage V tile transposed: vt[d][s_local]
      const int sl = threadIdx.x >> 3;           // 0..31
      const int dp = (threadIdx.x & 7) * 16;     // 0..112
      const _Float16* src = vbase + (size_t)(ks * 32 + sl) * kD + dp;
      #pragma unroll
      for (int j = 0; j < 16; ++j) vt[dp + j][sl] = src[j];
    }
    __syncthreads();
    const _Float16* prow = &sc[ml][ks * 32];
    v16h af = shuf16(*(const h8*)(prow + kb_a), *(const h8*)(prow + 16 + kb_a));
    const _Float16* vrow = &vt[n0 + ml][0];
    v16h bf = shuf16(*(const h8*)(vrow + kb_b), *(const h8*)(vrow + 16 + kb_b));
    oacc = wmma_f16(af, bf, oacc);
  }
  #pragma unroll
  for (int r = 0; r < 8; ++r) {
    const int m = r + rhi;
    const float v = oacc[r] / rsum[m];
    aoh[(size_t)(m0 + m) * (kH * kD) + h * kD + n0 + ml] = (_Float16)v;
  }
}

// ---------------------------------------------------------------------------
// Host-side launcher
// ---------------------------------------------------------------------------
extern "C" void kernel_launch(void* const* d_in, const int* in_sizes, int n_in,
                              void* d_out, int out_size, void* d_ws, size_t ws_size,
                              hipStream_t stream)
{
  const float* hidden = (const float*)d_in[0];
  const float* Wq     = (const float*)d_in[1];
  const float* Wkv    = (const float*)d_in[2];
  const float* Wo     = (const float*)d_in[3];
  const float* qnw    = (const float*)d_in[4];
  const float* knw    = (const float*)d_in[5];
  const float* cosb   = (const float*)d_in[6];
  const float* sinb   = (const float*)d_in[7];

  // workspace carve-up (~84 MB total)
  char* ws = (char*)d_ws;
  float*    qf  = (float*)ws;     ws += (size_t)kT * kHID * sizeof(float);        // 32 MB
  float*    kvf = (float*)ws;     ws += (size_t)kT * kKVH * kD * sizeof(float);   //  8 MB
  _Float16* qh  = (_Float16*)ws;  ws += (size_t)kH  * kT * kD * sizeof(_Float16); // 16 MB
  _Float16* kh  = (_Float16*)ws;  ws += (size_t)kKVH* kT * kD * sizeof(_Float16); //  4 MB
  _Float16* vh  = (_Float16*)ws;  ws += (size_t)kKVH* kT * kD * sizeof(_Float16); //  4 MB
  _Float16* aoh = (_Float16*)ws;  ws += (size_t)kT * kHID * sizeof(_Float16);     // 16 MB

  const dim3 blk(256);
  // 1) projections
  gemm_wmma<false><<<dim3(kHID / 128, kT / 128), blk, 0, stream>>>(
      hidden, Wq, qf, kT, kHID, kHID);
  gemm_wmma<false><<<dim3((kKVH * kD) / 128, kT / 128), blk, 0, stream>>>(
      hidden, Wkv, kvf, kT, kKVH * kD, kHID);
  // 2) norm + rope + relayout
  rmsnorm_rope<<<dim3(kT), blk, 0, stream>>>(qf, kvf, qnw, knw, cosb, sinb,
                                             qh, kh, vh);
  // 3) sparse attention
  attn_sparse<<<dim3(kT / 16, kH), blk, 0, stream>>>(qh, kh, vh, aoh);
  // 4) output projection
  gemm_wmma<true><<<dim3(kHID / 128, kT / 128), blk, 0, stream>>>(
      aoh, Wo, (float*)d_out, kT, kHID, kHID);
}
